// SpatialGNN_75471165325733
// MI455X (gfx1250) — matrix operand order repaired and checked by reading, not compiled, not run on hardware
//
#include <hip/hip_runtime.h>
#include <hip/hip_bf16.h>
#include <stdint.h>

typedef __attribute__((ext_vector_type(16))) _Float16 v16h;
typedef __attribute__((ext_vector_type(8)))  float    v8f;

// ---------------------------------------------------------------------------
// Fragment-packed layouts (CDNA5 16-bit WMMA, ISA 7.12.2):
//  A-pack: Ap[((mb*Kb + kb)*32 + lane)*16 + i]
//          = A[mb*16 + (lane&15)][kb*32 + ka(i, lane>>4)]
//          ka(i,h) = (i<8) ? 8h+i : 8+8h+i
//  B-pack: Bp[((nb*Kb + kb)*32 + lane)*16 + i]
//          = B[kb*32 + ka(i, lane>>4)][nb*16 + (lane&15)]
// Each v16h fragment is a 32-byte contiguous per-lane read (2x b128).
// ---------------------------------------------------------------------------

__device__ __forceinline__ int ka_of(int i, int half) {
    return (i < 8) ? (8 * half + i) : (8 + 8 * half + i);
}

// pack f32 activation [M,K] -> fragment-packed f16 (optional row mask -> 0)
__global__ __launch_bounds__(256)
void pack_a_kernel(const float* __restrict__ in, _Float16* __restrict__ out,
                   const uint8_t* __restrict__ mask, int M, int K)
{
    const int t = blockIdx.x * blockDim.x + threadIdx.x;
    if (t >= M * K) return;
    const int i    = t & 15;
    const int lane = (t >> 4) & 31;
    const int blk  = t >> 9;             // 512 elems per (mb,kb) block
    const int Kb   = K >> 5;
    const int kb   = blk % Kb;
    const int mb   = blk / Kb;
    const int row  = mb * 16 + (lane & 15);
    const int k    = kb * 32 + ka_of(i, lane >> 4);
    float v = in[(size_t)row * K + k];
    if (mask && mask[row]) v = 0.0f;
    out[t] = (_Float16)v;
}

// pack f32 weight [K,N] -> fragment-packed f16
__global__ __launch_bounds__(256)
void pack_b_kernel(const float* __restrict__ in, _Float16* __restrict__ out,
                   int K, int N)
{
    const int t = blockIdx.x * blockDim.x + threadIdx.x;
    if (t >= K * N) return;
    const int i    = t & 15;
    const int lane = (t >> 4) & 31;
    const int blk  = t >> 9;
    const int Kb   = K >> 5;
    const int kb   = blk % Kb;
    const int nb   = blk / Kb;
    const int k    = kb * 32 + ka_of(i, lane >> 4);
    const int n    = nb * 16 + (lane & 15);
    out[t] = (_Float16)in[(size_t)k * N + n];
}

// ---------------------------------------------------------------------------
// WMMA GEMM on fragment-packed operands.
// One wave computes a 32x32 tile (2x2 register blocking, 4 wmma / 4 loads).
// flags: bit0 = +bias, bit1 = relu, bit2 = +Cin
// Requires M%32==0, N%32==0, K%32==0 (true for all GEMMs here).
// ---------------------------------------------------------------------------
__global__ __launch_bounds__(256)
void gemm_wmma_packed(const _Float16* __restrict__ Ap, const _Float16* __restrict__ Bp,
                      const float* __restrict__ bias, const float* __restrict__ Cin,
                      float* __restrict__ C, int M, int N, int K, int flags)
{
    const int lane = threadIdx.x & 31;
    const int wave = threadIdx.x >> 5;
    const int Mt = M >> 5, Nt = N >> 5, Kb = K >> 5;
    const int t = blockIdx.x * 8 + wave;
    if (t >= Mt * Nt) return;
    const int mt = t % Mt;
    const int nt = t / Mt;

    const _Float16* a0p = Ap + ((size_t)(2 * mt)     * Kb * 32 + lane) * 16;
    const _Float16* a1p = Ap + ((size_t)(2 * mt + 1) * Kb * 32 + lane) * 16;
    const _Float16* b0p = Bp + ((size_t)(2 * nt)     * Kb * 32 + lane) * 16;
    const _Float16* b1p = Bp + ((size_t)(2 * nt + 1) * Kb * 32 + lane) * 16;

    v8f acc00 = {}, acc01 = {}, acc10 = {}, acc11 = {};

    #pragma unroll 2
    for (int kb = 0; kb < Kb; ++kb) {
        const size_t off = (size_t)kb * 512;   // 32 lanes * 16 f16
        const v16h a0 = *(const v16h*)(a0p + off);
        const v16h a1 = *(const v16h*)(a1p + off);
        const v16h b0 = *(const v16h*)(b0p + off);
        const v16h b1 = *(const v16h*)(b1p + off);
        acc00 = __builtin_amdgcn_wmma_f32_16x16x32_f16(false, a0, false, b0, (short)0, acc00, false, false);
        acc01 = __builtin_amdgcn_wmma_f32_16x16x32_f16(false, a0, false, b1, (short)0, acc01, false, false);
        acc10 = __builtin_amdgcn_wmma_f32_16x16x32_f16(false, a1, false, b0, (short)0, acc10, false, false);
        acc11 = __builtin_amdgcn_wmma_f32_16x16x32_f16(false, a1, false, b1, (short)0, acc11, false, false);
    }

    // Epilogue: C/D layout -> VGPR r: row = rb + r + 8*(lane>>4), col = cb + (lane&15)
    const int colh = lane & 15;
    const int rowh = 8 * (lane >> 4);
    auto store_tile = [&](const v8f& acc, int rb, int cb) {
        const int col = cb + colh;
        #pragma unroll
        for (int r = 0; r < 8; ++r) {
            const int row = rb + rowh + r;
            float v = acc[r];
            if (flags & 4) v += Cin[(size_t)row * N + col];
            if (flags & 1) v += bias[col];
            if (flags & 2) v = fmaxf(v, 0.0f);
            C[(size_t)row * N + col] = v;
        }
    };
    store_tile(acc00, mt * 32,      nt * 32);
    store_tile(acc01, mt * 32,      nt * 32 + 16);
    store_tile(acc10, mt * 32 + 16, nt * 32);
    store_tile(acc11, mt * 32 + 16, nt * 32 + 16);
}

// ---------------------------------------------------------------------------
__global__ __launch_bounds__(256)
void zero_f32_kernel(float* __restrict__ p, int n)
{
    int i = blockIdx.x * blockDim.x + threadIdx.x;
    if (i < n) p[i] = 0.0f;
}

// per-node attention partials: one wave per (n,h)
__global__ __launch_bounds__(256)
void att_partial_kernel(const float* __restrict__ h, const float* __restrict__ att,
                        float* __restrict__ ai, float* __restrict__ aj,
                        int NH, int H, int D)
{
    int idx = blockIdx.x * (blockDim.x >> 5) + (threadIdx.x >> 5);
    if (idx >= NH) return;
    const int n = idx / H, hh = idx % H;
    const int lane = threadIdx.x & 31;
    const float* hp  = h   + (size_t)n * H * D + (size_t)hh * D;
    const float* atp = att + (size_t)hh * 2 * D;
    float si = 0.0f, sj = 0.0f;
    for (int d = lane; d < D; d += 32) {
        const float hv = hp[d];
        si += hv * atp[d];
        sj += hv * atp[D + d];
    }
    #pragma unroll
    for (int off = 16; off; off >>= 1) {
        si += __shfl_down(si, off, 32);
        sj += __shfl_down(sj, off, 32);
    }
    if (lane == 0) { ai[idx] = si; aj[idx] = sj; }
}

// edge scatter: one wave per edge; per-edge head softmax then scaled scatter-add
__global__ __launch_bounds__(256)
void edge_kernel(const int* __restrict__ src, const int* __restrict__ dst,
                 int n_edges, int n_total,
                 const float* __restrict__ h,
                 const float* __restrict__ ai, const float* __restrict__ aj,
                 float* __restrict__ agg, int H, int dshift)
{
    const int e = blockIdx.x * (blockDim.x >> 5) + (threadIdx.x >> 5);
    if (e >= n_total) return;
    const int lane = threadIdx.x & 31;

    int s, t;
    if (e < n_edges) { s = src[e]; t = dst[e]; }
    else             { s = t = e - n_edges; }

    float alpha[8];
    float mx = -3.0e38f;
    for (int hh = 0; hh < H; ++hh) {
        float v = ai[(size_t)t * H + hh] + aj[(size_t)s * H + hh];
        v = (v > 0.0f) ? v : 0.2f * v;               // leaky_relu(0.2)
        alpha[hh] = v;
        mx = fmaxf(mx, v);
    }
    float sum = 0.0f;
    for (int hh = 0; hh < H; ++hh) { alpha[hh] = __expf(alpha[hh] - mx); sum += alpha[hh]; }
    const float inv = 1.0f / sum;

    const int HD = H << dshift;
    const float* hs = h   + (size_t)s * HD;
    float*       at = agg + (size_t)t * HD;
    for (int c = lane; c < HD; c += 32) {
        const float m = hs[c] * alpha[c >> dshift] * inv;
        __hip_atomic_fetch_add(&at[c], m, __ATOMIC_RELAXED, __HIP_MEMORY_SCOPE_AGENT);
    }
}

__global__ __launch_bounds__(256)
void final_mask_kernel(const float* __restrict__ tmp, const float* __restrict__ x,
                       const uint8_t* __restrict__ mask, float* __restrict__ out,
                       int total, int F)
{
    int i = blockIdx.x * blockDim.x + threadIdx.x;
    if (i >= total) return;
    out[i] = mask[i / F] ? tmp[i] : x[i];
}

// ---------------------------------------------------------------------------
static inline int cdiv(int a, int b) { return (a + b - 1) / b; }

extern "C" void kernel_launch(void* const* d_in, const int* in_sizes, int n_in,
                              void* d_out, int out_size, void* d_ws, size_t ws_size,
                              hipStream_t stream)
{
    const int F = 32;
    const int N = in_sizes[0] / F;       // 20000 nodes
    const int E = in_sizes[1] / 2;       // 200000 edges
    const int ET = E + N;                // + self loops

    const float*   x    = (const float*)d_in[0];
    const int*     ei   = (const int*)d_in[1];
    const uint8_t* mask = (const uint8_t*)d_in[2];
    const int* src = ei;
    const int* dst = ei + E;

    const float* W0   = (const float*)d_in[3];
    const float* att0 = (const float*)d_in[4];
    const float* Wo0  = (const float*)d_in[5];
    const float* bo0  = (const float*)d_in[6];
    const float* W1   = (const float*)d_in[7];
    const float* att1 = (const float*)d_in[8];
    const float* Wo1  = (const float*)d_in[9];
    const float* bo1  = (const float*)d_in[10];
    const float* W2   = (const float*)d_in[11];
    const float* att2 = (const float*)d_in[12];
    const float* Wo2  = (const float*)d_in[13];
    const float* bo2  = (const float*)d_in[14];
    const float* outW = (const float*)d_in[15];
    const float* outb = (const float*)d_in[16];
    const float* skW  = (const float*)d_in[17];
    const float* skb  = (const float*)d_in[18];

    char* wp = (char*)d_ws;
    auto alloc = [&](size_t bytes) -> void* {
        void* p = (void*)wp;
        wp += (bytes + 255) & ~(size_t)255;
        return p;
    };
    float*    h     = (float*)   alloc((size_t)N * 1024 * 4);
    float*    agg   = (float*)   alloc((size_t)N * 1024 * 4);
    _Float16* aggp  = (_Float16*)alloc((size_t)N * 1024 * 2);  // packed
    float*    xcur  = (float*)   alloc((size_t)N * 128 * 4);
    _Float16* xcurp = (_Float16*)alloc((size_t)N * 128 * 2);   // packed
    _Float16* xmp   = (_Float16*)alloc((size_t)N * F * 2);     // masked x, packed
    _Float16* xop   = (_Float16*)alloc((size_t)N * F * 2);     // original x, packed
    float*    ai    = (float*)   alloc((size_t)N * 8 * 4);
    float*    aj    = (float*)   alloc((size_t)N * 8 * 4);
    float*    tmpo  = (float*)   alloc((size_t)N * F * 4);
    _Float16* W0p   = (_Float16*)alloc(32 * 512 * 2);
    _Float16* Wo0p  = (_Float16*)alloc(512 * 64 * 2);
    _Float16* W1p   = (_Float16*)alloc(64 * 1024 * 2);
    _Float16* Wo1p  = (_Float16*)alloc(1024 * 128 * 2);
    _Float16* W2p   = (_Float16*)alloc(128 * 64 * 2);
    _Float16* Wo2p  = (_Float16*)alloc(64 * 64 * 2);
    _Float16* outWp = (_Float16*)alloc(64 * 32 * 2);
    _Float16* skWp  = (_Float16*)alloc(32 * 32 * 2);
    (void)ws_size; (void)n_in; (void)out_size;

    const dim3 blk(256);

    // activation packs (cast folded into fragment packing)
    pack_a_kernel<<<cdiv(N * F, 256), blk, 0, stream>>>(x, xmp, mask,    N, F);
    pack_a_kernel<<<cdiv(N * F, 256), blk, 0, stream>>>(x, xop, nullptr, N, F);

    // weight packs
    pack_b_kernel<<<cdiv(32 * 512, 256),   blk, 0, stream>>>(W0,  W0p,  32, 512);
    pack_b_kernel<<<cdiv(512 * 64, 256),   blk, 0, stream>>>(Wo0, Wo0p, 512, 64);
    pack_b_kernel<<<cdiv(64 * 1024, 256),  blk, 0, stream>>>(W1,  W1p,  64, 1024);
    pack_b_kernel<<<cdiv(1024 * 128, 256), blk, 0, stream>>>(Wo1, Wo1p, 1024, 128);
    pack_b_kernel<<<cdiv(128 * 64, 256),   blk, 0, stream>>>(W2,  W2p,  128, 64);
    pack_b_kernel<<<cdiv(64 * 64, 256),    blk, 0, stream>>>(Wo2, Wo2p, 64, 64);
    pack_b_kernel<<<cdiv(64 * 32, 256),    blk, 0, stream>>>(outW, outWp, 64, 32);
    pack_b_kernel<<<cdiv(32 * 32, 256),    blk, 0, stream>>>(skW,  skWp,  32, 32);

    struct LayerCfg { const _Float16 *Wp, *Wop; const float *att, *bo; int in, H, D, ds; };
    const LayerCfg layers[3] = {
        { W0p, Wo0p, att0, bo0, 32,  8, 64,  6 },
        { W1p, Wo1p, att1, bo1, 64,  8, 128, 7 },
        { W2p, Wo2p, att2, bo2, 128, 1, 64,  6 },
    };

    const _Float16* actp = xmp;
    for (int L = 0; L < 3; ++L) {
        const LayerCfg& c = layers[L];
        const int HD = c.H * c.D;

        // h = act @ W
        {
            const int tiles = (N >> 5) * (HD >> 5);
            gemm_wmma_packed<<<cdiv(tiles, 8), blk, 0, stream>>>(
                actp, c.Wp, nullptr, nullptr, h, N, HD, c.in, 0);
        }
        att_partial_kernel<<<cdiv(N * c.H, 8), blk, 0, stream>>>(
            h, c.att, ai, aj, N * c.H, c.H, c.D);
        zero_f32_kernel<<<cdiv(N * HD, 256), blk, 0, stream>>>(agg, N * HD);
        edge_kernel<<<cdiv(ET, 8), blk, 0, stream>>>(
            src, dst, E, ET, h, ai, aj, agg, c.H, c.ds);

        // xcur = relu(agg @ Wo + bo)
        pack_a_kernel<<<cdiv(N * HD, 256), blk, 0, stream>>>(agg, aggp, nullptr, N, HD);
        {
            const int tiles = (N >> 5) * (c.D >> 5);
            gemm_wmma_packed<<<cdiv(tiles, 8), blk, 0, stream>>>(
                aggp, c.Wop, c.bo, nullptr, xcur, N, c.D, HD, 1 | 2);
        }
        pack_a_kernel<<<cdiv(N * c.D, 256), blk, 0, stream>>>(xcur, xcurp, nullptr, N, c.D);
        actp = xcurp;
    }

    // tmpo = x3 @ out_W + out_b
    {
        const int tiles = (N >> 5) * (F >> 5);
        gemm_wmma_packed<<<cdiv(tiles, 8), blk, 0, stream>>>(
            xcurp, outWp, outb, nullptr, tmpo, N, F, 64, 1);
    }
    // tmpo += x_input @ skip_W + skip_b
    {
        const int tiles = (N >> 5) * (F >> 5);
        gemm_wmma_packed<<<cdiv(tiles, 8), blk, 0, stream>>>(
            xop, skWp, skb, tmpo, tmpo, N, F, 32, 1 | 4);
    }
    final_mask_kernel<<<cdiv(N * F, 256), blk, 0, stream>>>(
        tmpo, x, mask, (float*)d_out, N * F, F);
}
